// SelfAttention_3728031612959
// MI455X (gfx1250) — matrix operand order repaired and checked by reading, not compiled, not run on hardware
//
#include <hip/hip_runtime.h>
#include <hip/hip_bf16.h>
#include <math.h>

// Problem sizes (fixed by reference)
#define B_DIM 64
#define S_DIM 2048
#define U_DIM 1024

typedef __bf16 bf16_t;
typedef bf16_t v4bf  __attribute__((ext_vector_type(4)));
typedef bf16_t v8bf  __attribute__((ext_vector_type(8)));
typedef bf16_t v16bf __attribute__((ext_vector_type(16)));
typedef float  v8f   __attribute__((ext_vector_type(8)));

// ---- fused-scores kernel tiling ----
#define M_TILE     64          // rows (b*S+s) per block
#define N_CHUNK    128         // output cols per chunk (full K loop per chunk)
#define KB_CHUNK   128         // K depth staged per B buffer
#define A_STRIDE   1032        // bf16/row of A in LDS (1024 + 8 pad -> conflict-free frags)
#define BKF_STRIDE 132         // fp32/column of B in LDS (128 + 4 pad, 16B-aligned frags)
#define SMEM_A_BYTES   (M_TILE * A_STRIDE * 2)          // 132096
#define SMEM_BF_BYTES  (N_CHUNK * BKF_STRIDE * 4)       // 67584 per buffer
#define SMEM_B0_OFF    SMEM_A_BYTES                     // 132096
#define SMEM_B1_OFF    (SMEM_A_BYTES + SMEM_BF_BYTES)   // 199680
#define SMEM_SCORE_OFF (SMEM_A_BYTES + 2 * SMEM_BF_BYTES) // 267264
#define SMEM_TOTAL     (SMEM_SCORE_OFF + M_TILE * 4)    // 267520 < 320 KB WGP LDS

// fast tanh: v_exp_f32 + v_rcp_f32 instead of libm polynomial
__device__ __forceinline__ float fast_tanh(float x)
{
    const float e = __expf(-2.0f * fabsf(x));
    float t = (1.0f - e) * __builtin_amdgcn_rcpf(1.0f + e);
    return copysignf(t, x);
}

__device__ __forceinline__ void wait_async0()
{
    asm volatile("s_wait_asynccnt 0x0" ::: "memory");
}

// Issue async global->LDS copies for a 128K x 128N fp32 tile of Uw,
// transposed into n-major (k-contiguous) layout. Tracked by ASYNCcnt only:
// no VGPR round-trip, no loadcnt/dscnt interference with the WMMA stream.
__device__ __forceinline__ void stage_B_async(unsigned ldsOffBase,
                                              const float* __restrict__ Uw,
                                              int kbase, int nc, int tid)
{
    const int n  = tid & 127;            // this thread's column (coalesced per k)
    const int k0 = tid >> 7;             // 0 or 1; covers k = k0, k0+2, ...
    unsigned voff = (unsigned)((((kbase + k0) * U_DIM) + nc + n) * 4);     // global byte off
    unsigned loff = ldsOffBase + (unsigned)(n * (BKF_STRIDE * 4) + k0 * 4); // LDS byte off
    #pragma unroll
    for (int g = 0; g < 64; ++g) {
        asm volatile("global_load_async_to_lds_b32 %0, %1, %2"
                     :: "v"(loff), "v"(voff), "s"(Uw)
                     : "memory");
        voff += 2u * U_DIM * 4u;         // k += 2 (rows of Uw)
        loff += 8u;                      // 2 floats along the column
    }
}

// ---------------------------------------------------------------------------
// Kernel 1: W_s[b,n] = sum_u s_prev[b,u] * W_w[u,n] + W_b[n]   (fp32, tiny)
// ---------------------------------------------------------------------------
__global__ __launch_bounds__(256) void ws_kernel(
    const float* __restrict__ s_prev, const float* __restrict__ Ww,
    const float* __restrict__ Wb, float* __restrict__ Ws)
{
    const int b = blockIdx.y;
    const int n = blockIdx.x * 256 + threadIdx.x;
    const float* sp = s_prev + b * U_DIM;
    float acc = Wb[n];
    #pragma unroll 8
    for (int u = 0; u < U_DIM; ++u)
        acc = fmaf(sp[u], Ww[(size_t)u * U_DIM + n], acc);
    Ws[b * U_DIM + n] = acc;
}

// ---------------------------------------------------------------------------
// Kernel 2: fused U_h GEMM (bf16 WMMA) + tanh(Ws + U_h + Ub) . Vw -> scores
// One block = 8 wave32 = 64 rows x full 1024 cols.
// B tiles stream via async-to-LDS (double-buffered), overlapped with WMMA.
// ---------------------------------------------------------------------------
__global__ __launch_bounds__(256) void attn_scores_kernel(
    const float* __restrict__ hidden, const float* __restrict__ Uw,
    const float* __restrict__ Ub, const float* __restrict__ Ws,
    const float* __restrict__ Vw, const float* __restrict__ Vb,
    float* __restrict__ scores)
{
    extern __shared__ char smem[];
    bf16_t* ldsA     = (bf16_t*)smem;                     // [64][A_STRIDE] bf16
    float*  ldsB0f   = (float*)(smem + SMEM_B0_OFF);      // [128][BKF_STRIDE] fp32 n-major
    float*  ldsB1f   = (float*)(smem + SMEM_B1_OFF);
    float*  ldsScore = (float*)(smem + SMEM_SCORE_OFF);   // [64]

    const int tid  = threadIdx.x;
    const int lane = tid & 31;
    const int wid  = tid >> 5;
    const int mt   = wid & 3;       // wave's M-tile (rows mt*16..+15)
    const int nq   = wid >> 2;      // wave's N half within the chunk (64 cols)
    const int r    = lane & 15;
    const int h    = lane >> 4;

    const int rowBase = blockIdx.x * M_TILE;    // global row = b*S + s
    const int bIdx    = rowBase >> 11;          // rowBase / S_DIM (all rows share b)

    if (tid < M_TILE) ldsScore[tid] = 0.0f;

    // ---- stage A panel: 64 rows x 1024 K of hidden -> bf16 in LDS (once) ----
    {
        const float4* aSrc = (const float4*)(hidden + (size_t)rowBase * U_DIM);
        #pragma unroll 4
        for (int i = 0; i < 64; ++i) {
            const int gidx = tid + i * 256;          // float4 index within panel
            const float4 f = aSrc[gidx];
            const int m = gidx >> 8;                 // (gidx*4)/1024
            const int k = (gidx & 255) * 4;
            v4bf p = { (bf16_t)f.x, (bf16_t)f.y, (bf16_t)f.z, (bf16_t)f.w };
            *(v4bf*)(ldsA + m * A_STRIDE + k) = p;
        }
    }

    const float* WsRow = Ws + bIdx * U_DIM;
    const bf16_t* aRow = ldsA + (mt * 16 + r) * A_STRIDE + h * 8;

    for (int nc = 0; nc < U_DIM; nc += N_CHUNK) {
        v8f acc[4];
        #pragma unroll
        for (int i = 0; i < 4; ++i) acc[i] = (v8f){0.f,0.f,0.f,0.f,0.f,0.f,0.f,0.f};

        stage_B_async(SMEM_B0_OFF, Uw, 0, nc, tid);   // pre-issue first K chunk

        for (int kb = 0; kb < U_DIM / KB_CHUNK; ++kb) {
            const float* curB = (kb & 1) ? ldsB1f : ldsB0f;

            wait_async0();                // current buffer's async copies landed
            __syncthreads();              // publish to all waves; next buffer free

            if (kb + 1 < U_DIM / KB_CHUNK)   // overlap next chunk with WMMA below
                stage_B_async((kb & 1) ? SMEM_B0_OFF : SMEM_B1_OFF,
                              Uw, (kb + 1) * KB_CHUNK, nc, tid);

            // ---- 4 K-steps of 32 from this buffer ----
            #pragma unroll
            for (int ks = 0; ks < 4; ++ks) {
                // A fragment (ISA 7.12.2 16-bit A 16x32 layout), already bf16
                const bf16_t* ap = aRow + kb * KB_CHUNK + ks * 32;
                const v8bf a0 = *(const v8bf*)ap;          // K = h*8 .. +7
                const v8bf a1 = *(const v8bf*)(ap + 16);   // K = 16+h*8 .. +7
                const v16bf av = __builtin_shufflevector(a0, a1,
                    0,1,2,3,4,5,6,7,8,9,10,11,12,13,14,15);

                // load all 4 B fragments (fp32 -> bf16 cvt co-executes w/ WMMA)
                v16bf bv[4];
                #pragma unroll
                for (int i = 0; i < 4; ++i) {
                    const float* bp = curB + (nq * 64 + i * 16 + r) * BKF_STRIDE
                                           + ks * 32 + h * 8;
                    const float4 f0 = ((const float4*)bp)[0];
                    const float4 f1 = ((const float4*)bp)[1];
                    const float4 f2 = ((const float4*)bp)[4];
                    const float4 f3 = ((const float4*)bp)[5];
                    bv[i] = (v16bf){ (bf16_t)f0.x,(bf16_t)f0.y,(bf16_t)f0.z,(bf16_t)f0.w,
                                     (bf16_t)f1.x,(bf16_t)f1.y,(bf16_t)f1.z,(bf16_t)f1.w,
                                     (bf16_t)f2.x,(bf16_t)f2.y,(bf16_t)f2.z,(bf16_t)f2.w,
                                     (bf16_t)f3.x,(bf16_t)f3.y,(bf16_t)f3.z,(bf16_t)f3.w };
                }
                #pragma unroll
                for (int i = 0; i < 4; ++i)
                    acc[i] = __builtin_amdgcn_wmma_f32_16x16x32_bf16(
                        false, av, false, bv[i], (short)0, acc[i], false, false);
            }
        }

        // ---- epilogue: partial score += sum_n tanh(acc + Ub + Ws) * Vw ----
        float sv[8];
        #pragma unroll
        for (int v = 0; v < 8; ++v) sv[v] = 0.f;
        #pragma unroll
        for (int i = 0; i < 4; ++i) {
            const int n = nc + nq * 64 + i * 16 + r;   // this lane's column
            const float uws = Ub[n] + WsRow[n];        // hoisted out of v-loop
            const float vw  = Vw[n];
            #pragma unroll
            for (int v = 0; v < 8; ++v)
                sv[v] = fmaf(fast_tanh(acc[i][v] + uws), vw, sv[v]);
        }
        #pragma unroll
        for (int v = 0; v < 8; ++v) {
            float s = sv[v];
            s += __shfl_xor(s, 1, 32);     // reduce 16 lanes of this half
            s += __shfl_xor(s, 2, 32);
            s += __shfl_xor(s, 4, 32);
            s += __shfl_xor(s, 8, 32);
            if (r == 0)
                atomicAdd(&ldsScore[mt * 16 + v + 8 * h], s);
        }
    }

    __syncthreads();
    if (tid < M_TILE)
        scores[rowBase + tid] = ldsScore[tid] + Vb[0];
}

// ---------------------------------------------------------------------------
// Kernel 3: softmax over S per batch -> weights (written straight into d_out)
// ---------------------------------------------------------------------------
__global__ __launch_bounds__(256) void softmax_kernel(
    const float* __restrict__ scores, float* __restrict__ weights)
{
    __shared__ float red[8];
    const int b = blockIdx.x, tid = threadIdx.x;
    const int lane = tid & 31, wid = tid >> 5;
    const float* sc = scores + b * S_DIM;

    float v[8];
    float mx = -INFINITY;
    #pragma unroll
    for (int j = 0; j < 8; ++j) { v[j] = sc[tid + j * 256]; mx = fmaxf(mx, v[j]); }
    #pragma unroll
    for (int o = 16; o >= 1; o >>= 1) mx = fmaxf(mx, __shfl_xor(mx, o, 32));
    if (lane == 0) red[wid] = mx;
    __syncthreads();
    mx = red[0];
    #pragma unroll
    for (int w = 1; w < 8; ++w) mx = fmaxf(mx, red[w]);

    float sum = 0.f;
    #pragma unroll
    for (int j = 0; j < 8; ++j) { v[j] = __expf(v[j] - mx); sum += v[j]; }
    #pragma unroll
    for (int o = 16; o >= 1; o >>= 1) sum += __shfl_xor(sum, o, 32);
    __syncthreads();
    if (lane == 0) red[wid] = sum;
    __syncthreads();
    sum = 0.f;
    #pragma unroll
    for (int w = 0; w < 8; ++w) sum += red[w];

    const float inv = 1.0f / sum;
    #pragma unroll
    for (int j = 0; j < 8; ++j)
        weights[b * S_DIM + tid + j * 256] = v[j] * inv;
}

// ---------------------------------------------------------------------------
// Kernel 4: context[b,n] = sum_s weights[b,s] * hidden[b,s,n]  (HBM pass 2)
// ---------------------------------------------------------------------------
__global__ __launch_bounds__(256) void context_kernel(
    const float* __restrict__ hidden, const float* __restrict__ weights,
    float* __restrict__ context)
{
    __shared__ float sw[S_DIM];
    const int b = blockIdx.y;
    const int n = blockIdx.x * 256 + threadIdx.x;
    #pragma unroll
    for (int j = 0; j < 8; ++j)
        sw[threadIdx.x + j * 256] = weights[b * S_DIM + threadIdx.x + j * 256];
    __syncthreads();

    const float* hb = hidden + (size_t)b * S_DIM * U_DIM + n;
    float acc = 0.f;
    #pragma unroll 8
    for (int s = 0; s < S_DIM; ++s)
        acc = fmaf(sw[s], hb[(size_t)s * U_DIM], acc);
    context[b * U_DIM + n] = acc;
}

// ---------------------------------------------------------------------------
extern "C" void kernel_launch(void* const* d_in, const int* in_sizes, int n_in,
                              void* d_out, int out_size, void* d_ws, size_t ws_size,
                              hipStream_t stream)
{
    const float* s_prev = (const float*)d_in[0];
    const float* hidden = (const float*)d_in[1];
    const float* Ww     = (const float*)d_in[2];
    const float* Wb     = (const float*)d_in[3];
    const float* Uw     = (const float*)d_in[4];
    const float* Ub     = (const float*)d_in[5];
    const float* Vw     = (const float*)d_in[6];
    const float* Vb     = (const float*)d_in[7];

    float* context = (float*)d_out;                      // [64, 1024]
    float* weights = (float*)d_out + B_DIM * U_DIM;      // [64, 2048]
    float* Ws      = (float*)d_ws;                       // [64, 1024] scratch
    float* scores  = (float*)d_ws + B_DIM * U_DIM;       // [64, 2048] scratch

    ws_kernel<<<dim3(U_DIM / 256, B_DIM), 256, 0, stream>>>(s_prev, Ww, Wb, Ws);

    attn_scores_kernel<<<(B_DIM * S_DIM) / M_TILE, 256, SMEM_TOTAL, stream>>>(
        hidden, Uw, Ub, Ws, Vw, Vb, scores);

    softmax_kernel<<<B_DIM, 256, 0, stream>>>(scores, weights);

    context_kernel<<<dim3(U_DIM / 256, B_DIM), 256, 0, stream>>>(hidden, weights, context);
}
